// MultiCategoricalActor_7868380086274
// MI455X (gfx1250) — matrix operand order repaired and checked by reading, not compiled, not run on hardware
//
#include <hip/hip_runtime.h>
#include <hip/hip_bf16.h>
#include <math.h>

#define B_SZ   2048
#define ENB    16
#define RBG    32
#define MAXREQ 50
#define OBS_F  400
#define H_DIM  1024
#define OUT_NN (ENB * RBG * (MAXREQ + 1))          // 26112
#define IN_RAW (OBS_F + ENB * RBG + ENB * MAXREQ)  // 1712
#define IN_PAD 1728                                // next multiple of 32

typedef __attribute__((ext_vector_type(16))) __bf16 bf16x16;
typedef __attribute__((ext_vector_type(8)))  float  f32x8;

struct U4x2 { uint4 a, b; };

__device__ __forceinline__ unsigned short f32_to_bf16(float f) {
  unsigned int u = __float_as_uint(f);
  u += 0x7FFFu + ((u >> 16) & 1u);   // round-to-nearest-even
  return (unsigned short)(u >> 16);
}

// Low 32 bits of a generic LDS pointer are the LDS byte address
__device__ __forceinline__ unsigned lds_addr32(const void* p) {
  return (unsigned)(unsigned long long)p;
}

// CDNA5 async DMA: 16B global -> LDS, tracked by ASYNCcnt.
__device__ __forceinline__ void async_copy_b128(unsigned lds_dst, const void* gsrc) {
  asm volatile("global_load_async_to_lds_b128 %0, %1, off"
               :: "v"(lds_dst), "v"((unsigned long long)gsrc) : "memory");
}

__device__ __forceinline__ void wait_async0() {
#if __has_builtin(__builtin_amdgcn_s_wait_asynccnt)
  __builtin_amdgcn_s_wait_asynccnt(0);
#else
  asm volatile("s_wait_asynccnt 0x0" ::: "memory");
#endif
}

// Build bf16 input: concat(obs[400], rbgMap[512], invFlag[800]) padded to 1728.
__global__ __launch_bounds__(256) void pack_input_kernel(
    const float* __restrict__ obs, const float* __restrict__ rbg,
    const float* __restrict__ inv, unsigned short* __restrict__ out)
{
  int idx = blockIdx.x * 256 + threadIdx.x;
  if (idx >= B_SZ * IN_PAD) return;
  int b = idx / IN_PAD;
  int c = idx - b * IN_PAD;
  float v;
  if (c < OBS_F)                    v = obs[b * OBS_F + c];
  else if (c < OBS_F + ENB * RBG)   v = rbg[b * (ENB * RBG) + (c - OBS_F)];
  else if (c < IN_RAW)              v = inv[b * (ENB * MAXREQ) + (c - (OBS_F + ENB * RBG))];
  else                              v = 0.0f;
  out[idx] = f32_to_bf16(v);
}

// Tiled GEMM: C[M,N] = act(A[M,KPAD](bf16) * W[KREAL,N](f32->bf16) + bias)
// 256 threads = 8 wave32, 128x128 tile, BK=32, v_wmma_f32_16x16x32_bf16.
// Double-buffered LDS; A staged via global_load_async_to_lds_b128 (ASYNCcnt),
// B staged through registers (f32->bf16 convert) overlapping compute.
// MODE 0: tanh -> bf16. MODE 1: fused action mask -> f32.
template <int MODE, int KPAD, int KREAL, int N>
__global__ __launch_bounds__(256) void mlp_gemm_kernel(
    const unsigned short* __restrict__ A,
    const float* __restrict__ W,
    const float* __restrict__ bias,
    unsigned short* __restrict__ Hout,
    float* __restrict__ Fout,
    const float* __restrict__ rbgMap,
    const float* __restrict__ invFlag)
{
  constexpr int NSTEP = KPAD / 32;
  // Row stride 40 ushorts (80B) to stagger LDS banks. 2 buffers x 10KB each.
  __shared__ __align__(16) unsigned short Asm[2][128 * 40]; // A tile [m][k]
  __shared__ __align__(16) unsigned short Bsm[2][128 * 40]; // B tile transposed [n][k]

  const int tid = threadIdx.x;
  const int m0 = blockIdx.y * 128;
  const int n0 = blockIdx.x * 128;

  const int wid  = tid >> 5;       // wave32
  const int lane = tid & 31;
  const int hf   = lane >> 4;      // lane half (0/1)
  const int ln   = lane & 15;
  const int wm   = wid >> 2;       // 0..1 -> 64 rows per wave
  const int wn   = wid & 3;        // 0..3 -> 32 cols per wave

  f32x8 acc[4][2];
  #pragma unroll
  for (int i = 0; i < 4; ++i)
    #pragma unroll
    for (int j = 0; j < 2; ++j)
      acc[i][j] = (f32x8)0.0f;

  float breg[4][4];   // W staging registers (per-thread 16 f32)

  // Issue async A-tile copy for K-step kt into buffer `buf` (2 x b128/thread).
  auto asyncA = [&](int kt, int buf) {
    #pragma unroll
    for (int i = 0; i < 2; ++i) {
      int id  = tid + i * 256;          // 0..511
      int row = id >> 2;
      int q   = id & 3;                 // 8-ushort (16B) chunk
      async_copy_b128(lds_addr32(&Asm[buf][row * 40 + q * 8]),
                      A + (size_t)(m0 + row) * KPAD + kt + q * 8);
    }
  };
  // Load W tile for K-step kt into registers (coalesced across lanes, column
  // strided by N with compile-time immediate offsets).
  auto loadB = [&](int kt) {
    #pragma unroll
    for (int i = 0; i < 4; ++i) {
      int id = tid + i * 256;           // 0..1023
      int n  = id & 127;
      int kq = id >> 7;                 // 0..7 (k-quad)
      int kg = kt + kq * 4;
      const float* p = W + (size_t)((unsigned)kg * N + n0 + n);
      #pragma unroll
      for (int kk = 0; kk < 4; ++kk) {
        if (KPAD == KREAL) breg[i][kk] = p[(size_t)kk * N];
        else breg[i][kk] = (kg + kk < KREAL) ? p[(size_t)kk * N] : 0.0f;
      }
    }
  };
  // Convert + store W registers into LDS transposed [n][k].
  auto storeB = [&](int buf) {
    #pragma unroll
    for (int i = 0; i < 4; ++i) {
      int id = tid + i * 256;
      int n  = id & 127;
      int kq = id >> 7;
      uint2 pk;
      pk.x = (unsigned)f32_to_bf16(breg[i][0]) | ((unsigned)f32_to_bf16(breg[i][1]) << 16);
      pk.y = (unsigned)f32_to_bf16(breg[i][2]) | ((unsigned)f32_to_bf16(breg[i][3]) << 16);
      *reinterpret_cast<uint2*>(&Bsm[buf][n * 40 + kq * 4]) = pk;
    }
  };

  // ---- software pipeline prologue ----
  loadB(0);
  asyncA(0, 0);

  for (int it = 0; it < NSTEP; ++it) {
    const int buf = it & 1;
    storeB(buf);      // consumes loads issued last iteration (latency hidden)
    wait_async0();    // own async writes to Asm[buf] complete
    __syncthreads();  // everyone's tile `buf` visible; buf^1 free to overwrite
    if (it + 1 < NSTEP) {
      loadB((it + 1) * 32);
      asyncA((it + 1) * 32, buf ^ 1);
    }

    // --- Fragment loads per documented CDNA5 VGPR layouts ---
    bf16x16 afrag[4], bfrag[2];
    #pragma unroll
    for (int fm = 0; fm < 4; ++fm) {
      // A 16x32: lanes 0-15 hold K {0..7,16..23}; lanes 16-31 hold K {8..15,24..31}
      const int am = wm * 64 + fm * 16 + ln;
      U4x2 t;
      t.a = *reinterpret_cast<const uint4*>(&Asm[buf][am * 40 + hf * 8]);
      t.b = *reinterpret_cast<const uint4*>(&Asm[buf][am * 40 + hf * 8 + 16]);
      afrag[fm] = __builtin_bit_cast(bf16x16, t);
    }
    #pragma unroll
    for (int fn = 0; fn < 2; ++fn) {
      // B 32x16: lanes 0-15 col=lane K {0..7,16..23}; lanes 16-31 col=lane-16 K {8..15,24..31}
      const int bn = wn * 32 + fn * 16 + ln;
      U4x2 t;
      t.a = *reinterpret_cast<const uint4*>(&Bsm[buf][bn * 40 + hf * 8]);
      t.b = *reinterpret_cast<const uint4*>(&Bsm[buf][bn * 40 + hf * 8 + 16]);
      bfrag[fn] = __builtin_bit_cast(bf16x16, t);
    }
    #pragma unroll
    for (int fm = 0; fm < 4; ++fm)
      #pragma unroll
      for (int fn = 0; fn < 2; ++fn)
        acc[fm][fn] = __builtin_amdgcn_wmma_f32_16x16x32_bf16(
            false, afrag[fm], false, bfrag[fn], (short)0, acc[fm][fn],
            false, false);
  }

  // --- Epilogue (C/D layout: vgpr j -> row j + 8*lanehalf, col = lane&15) ---
  #pragma unroll
  for (int fm = 0; fm < 4; ++fm) {
    #pragma unroll
    for (int fn = 0; fn < 2; ++fn) {
      const int col = n0 + wn * 32 + fn * 16 + ln;
      const float bv = bias[col];
      if (MODE == 0) {
        #pragma unroll
        for (int j = 0; j < 8; ++j) {
          const int row = m0 + wm * 64 + fm * 16 + hf * 8 + j;
          const float v = tanhf(acc[fm][fn][j] + bv);
          Hout[(size_t)row * N + col] = f32_to_bf16(v);
        }
      } else {
        const int er = col / 51;         // ENB*RBG index
        const int q  = col - er * 51;    // 0..50
        const int e  = er >> 5;          // ENB index
        #pragma unroll
        for (int j = 0; j < 8; ++j) {
          const int row = m0 + wm * 64 + fm * 16 + hf * 8 + j;
          const float v = acc[fm][fn][j] + bv;
          bool masked = false;
          if (q != 0) {
            const float iv = invFlag[(size_t)row * (ENB * MAXREQ) + e * MAXREQ + (q - 1)];
            if (iv == -1.0f)
              masked = (rbgMap[(size_t)row * (ENB * RBG) + er] == 0.0f);
            else
              masked = (iv != 0.0f);
          }
          Fout[(size_t)row * OUT_NN + col] = masked ? -__builtin_inff() : v;
        }
      }
    }
  }
}

extern "C" void kernel_launch(void* const* d_in, const int* in_sizes, int n_in,
                              void* d_out, int out_size, void* d_ws, size_t ws_size,
                              hipStream_t stream)
{
  const float* obs = (const float*)d_in[0];
  const float* rbg = (const float*)d_in[1];
  const float* inv = (const float*)d_in[2];
  const float* W1  = (const float*)d_in[3];
  const float* b1  = (const float*)d_in[4];
  const float* W2  = (const float*)d_in[5];
  const float* b2  = (const float*)d_in[6];
  const float* W3  = (const float*)d_in[7];
  const float* b3  = (const float*)d_in[8];

  unsigned short* inp = (unsigned short*)d_ws;                 // 2048 x 1728 bf16
  unsigned short* h1  = inp + (size_t)B_SZ * IN_PAD;           // 2048 x 1024 bf16
  unsigned short* h2  = h1 + (size_t)B_SZ * H_DIM;             // 2048 x 1024 bf16
  float* out = (float*)d_out;                                  // 2048 x 26112 f32

  pack_input_kernel<<<(B_SZ * IN_PAD) / 256, 256, 0, stream>>>(obs, rbg, inv, inp);

  mlp_gemm_kernel<0, IN_PAD, IN_RAW, H_DIM>
      <<<dim3(H_DIM / 128, B_SZ / 128), 256, 0, stream>>>(
      inp, W1, b1, h1, nullptr, nullptr, nullptr);

  mlp_gemm_kernel<0, H_DIM, H_DIM, H_DIM>
      <<<dim3(H_DIM / 128, B_SZ / 128), 256, 0, stream>>>(
      h1, W2, b2, h2, nullptr, nullptr, nullptr);

  mlp_gemm_kernel<1, H_DIM, H_DIM, OUT_NN>
      <<<dim3(OUT_NN / 128, B_SZ / 128), 256, 0, stream>>>(
      h2, W3, b3, nullptr, out, rbg, inv);
}
